// VQVAE_62586263437866
// MI455X (gfx1250) — matrix operand order repaired and checked by reading, not compiled, not run on hardware
//
#include <hip/hip_runtime.h>
#include <hip/hip_bf16.h>

typedef __attribute__((ext_vector_type(16))) _Float16 v16h;
typedef __attribute__((ext_vector_type(8)))  _Float16 v8h;
typedef __attribute__((ext_vector_type(8)))  float    v8f;

#define FLAG_RELU_IN    1
#define FLAG_RELU_OUT   2
#define FLAG_TRANSPOSE  4
#define FLAG_RESID      8
#define FLAG_IN_NCHW32  16
#define FLAG_OUT_NCHW32 32

union H8u { v8h v; _Float16 h[8]; };

__device__ __forceinline__ int iclamp(int v, int lo, int hi) {
    return v < lo ? lo : (v > hi ? hi : v);
}

__device__ __forceinline__ v8h relu8(v8h x) {
    H8u u; u.v = x;
    #pragma unroll
    for (int j = 0; j < 8; ++j)
        u.h[j] = u.h[j] > (_Float16)0 ? u.h[j] : (_Float16)0;
    return u.v;
}

// ---------------------------------------------------------------------------
// Weight prep: f32 OIHW (conv) or IOHW (tconv) -> f16 [oc][kh][kw][ic]
// matching the GEMM K-order (kh, kw, ic).
// ---------------------------------------------------------------------------
template<int IC, int OC, int KH, int KW, bool TR>
__global__ __launch_bounds__(256) void prep_w(
    const float* __restrict__ w, _Float16* __restrict__ wh)
{
    constexpr int N = OC * KH * KW * IC;
    const int e = blockIdx.x * 256 + threadIdx.x;
    if (e < N) {
        const int ic = e % IC;  int t = e / IC;
        const int kw = t % KW;  t /= KW;
        const int kh = t % KH;  const int oc = t / KH;
        const float v = TR ? w[((ic * OC + oc) * KH + kh) * KW + kw]
                           : w[((oc * IC + ic) * KH + kh) * KW + kw];
        wh[e] = (_Float16)v;
    }
}

// ---------------------------------------------------------------------------
// Implicit-GEMM conv / transposed conv via V_WMMA_F32_16X16X32_F16.
// Activations: NHWC f16 (except flagged NCHW f32 boundary layers).
// Weights: f16 [oc][kh][kw][ic]; A fragments loaded directly from global
// (no LDS stage). B tile staged in LDS with vector loads/stores.
// Block: 256 thr = 8 waves; tile = 128 OC x 128 positions; 8 WMMAs/wave/chunk.
// ---------------------------------------------------------------------------
template<int IC, int OC, int IH, int IW, int OH, int OW,
         int KH, int KW, int STRIDE, int PAD, int FLAGS>
__global__ __launch_bounds__(256) void conv_wmma(
    const void* __restrict__ xin, const _Float16* __restrict__ wh,
    const float* __restrict__ bias, const _Float16* __restrict__ res16,
    void* __restrict__ yout)
{
    constexpr int  Kdim = KH * KW * IC;
    constexpr int  OHW  = OH * OW;
    constexpr bool TR   = (FLAGS & FLAG_TRANSPOSE) != 0;
    constexpr bool RIN  = (FLAGS & FLAG_RELU_IN) != 0;
    constexpr bool ROUT = (FLAGS & FLAG_RELU_OUT) != 0;
    constexpr bool RES  = (FLAGS & FLAG_RESID) != 0;
    constexpr bool IN32 = (FLAGS & FLAG_IN_NCHW32) != 0;
    constexpr bool OUT32= (FLAGS & FLAG_OUT_NCHW32) != 0;

    __shared__ __align__(32) _Float16 sB[128 * 32]; // [n][kk]

    const int tid  = threadIdx.x;
    const int lane = tid & 31;
    const int wave = tid >> 5;
    const int hi   = lane >> 4;
    const int lm   = lane & 15;
    const int posBase = blockIdx.x * 128;

    const int nB   = tid >> 1;
    const int kkB  = (tid & 1) * 16;
    const int posB = posBase + nB;
    const int bB   = posB / OHW;
    const int rrB  = posB - bB * OHW;
    const int ohB  = rrB / OW;
    const int owB  = rrB - ohB * OW;

    const _Float16* x16 = (const _Float16*)xin;
    const float*    x32 = (const float*)xin;

    const int arow  = wave * 16 + lm;
    const int arowc = (arow < OC) ? arow : (OC - 1);   // clamp (OC=3 case)
    const _Float16* wr = wh + (size_t)arowc * Kdim;

    const v8h zero8 = {};
    v8f acc[8] = {{}, {}, {}, {}, {}, {}, {}, {}};

    for (int kb = 0; kb < Kdim; kb += 32) {
        __syncthreads();
        // ================= stage B (im2col tile) =================
        if (IN32) {
            // boundary layer: gather from NCHW f32 input (IC=3)
            float bv[16];
            #pragma unroll
            for (int j = 0; j < 16; ++j) {
                const int kk   = kb + kkB + j;
                const int kkc  = (kk < Kdim) ? kk : (Kdim - 1);
                const int khkw = kkc / IC;
                const int ic   = kkc - khkw * IC;
                const int kh   = khkw / KW;
                const int kw   = khkw - kh * KW;
                const int ih   = ohB * STRIDE - PAD + kh;
                const int iw   = owB * STRIDE - PAD + kw;
                const bool valid = (kk < Kdim) && ih >= 0 && ih < IH &&
                                   iw >= 0 && iw < IW;
                const int ihc = iclamp(ih, 0, IH - 1);
                const int iwc = iclamp(iw, 0, IW - 1);
                float v = x32[((bB * IC + ic) * IH + ihc) * IW + iwc];
                bv[j] = valid ? v : 0.0f;
            }
            H8u p0, p1;
            #pragma unroll
            for (int j = 0; j < 8; ++j) {
                p0.h[j] = (_Float16)bv[j];
                p1.h[j] = (_Float16)bv[8 + j];
            }
            *(v8h*)(sB + nB * 32 + kkB)     = p0.v;
            *(v8h*)(sB + nB * 32 + kkB + 8) = p1.v;
        } else {
            // NHWC f16: the 16-K run lies in one (kh,kw) tap -> 2 vector loads
            const int kkstart = kb + kkB;
            const int khkw = kkstart / IC;
            const int ic0  = kkstart - khkw * IC;
            const int kh   = khkw / KW;
            const int kw   = khkw - kh * KW;
            bool valid;
            int ih, iw;
            if (!TR) {
                ih = ohB * STRIDE - PAD + kh;
                iw = owB * STRIDE - PAD + kw;
                valid = ih >= 0 && ih < IH && iw >= 0 && iw < IW;
            } else {
                const int th = ohB + PAD - kh;
                const int tw = owB + PAD - kw;
                valid = th >= 0 && tw >= 0 &&
                        (th % STRIDE) == 0 && (tw % STRIDE) == 0;
                ih = th / STRIDE;
                iw = tw / STRIDE;
                valid = valid && ih < IH && iw < IW;
            }
            const int ihc = iclamp(ih, 0, IH - 1);
            const int iwc = iclamp(iw, 0, IW - 1);
            const _Float16* p =
                x16 + (((size_t)(bB * IH + ihc) * IW + iwc) * IC + ic0);
            v8h b0 = *(const v8h*)p;
            v8h b1 = *(const v8h*)(p + 8);
            if (!valid) { b0 = zero8; b1 = zero8; }
            if (RIN) { b0 = relu8(b0); b1 = relu8(b1); }
            *(v8h*)(sB + nB * 32 + kkB)     = b0;
            *(v8h*)(sB + nB * 32 + kkB + 8) = b1;
        }
        __syncthreads();

        // speculative prefetch of next weight chunk (global_prefetch_b8)
        if (kb + 32 < Kdim)
            __builtin_prefetch(wr + kb + 32, 0, 1);

        // ---- A fragment: 2 direct 16B global loads (contiguous k = 16*hi + s) ----
        const int s0 = kb + hi * 16;
        const int s1 = kb + hi * 16 + 8;
        v8h a0 = *(const v8h*)(wr + ((s0 + 8 <= Kdim) ? s0 : 0));
        v8h a1 = *(const v8h*)(wr + ((s1 + 8 <= Kdim) ? s1 : 0));
        if (s0 + 8 > Kdim) a0 = zero8;
        if (s1 + 8 > Kdim) a1 = zero8;
        union { v16h v; v8h h2[2]; } ua;
        ua.h2[0] = a0;
        ua.h2[1] = a1;

        #pragma unroll
        for (int nt = 0; nt < 8; ++nt) {
            const v16h bf = *(const v16h*)(sB + (nt * 16 + lm) * 32 + hi * 16);
            acc[nt] = __builtin_amdgcn_wmma_f32_16x16x32_f16(
                false, ua.v, false, bf, (short)0, acc[nt], false, false);
        }
    }

    // ================= epilogue =================
    const int ocBase = wave * 16 + hi * 8;
    #pragma unroll
    for (int nt = 0; nt < 8; ++nt) {
        const int pos = posBase + nt * 16 + lm;
        H8u rv;
        if (RES) rv.v = *(const v8h*)(res16 + (size_t)pos * OC + ocBase);
        if (OUT32) {
            const int b  = pos / OHW;
            const int rr = pos - b * OHW;
            const int oh = rr / OW;
            const int ow = rr - oh * OW;
            float* y32 = (float*)yout;
            #pragma unroll
            for (int v = 0; v < 8; ++v) {
                const int oc = ocBase + v;
                if (oc < OC) {
                    float r = acc[nt][v] + bias[oc];
                    if (ROUT) r = fmaxf(r, 0.0f);
                    if (RES)  r += (float)rv.h[v];
                    y32[((b * OC + oc) * OH + oh) * OW + ow] = r;
                }
            }
        } else {
            H8u o;
            #pragma unroll
            for (int v = 0; v < 8; ++v) {
                float r = acc[nt][v] + bias[ocBase + v];
                if (ROUT) r = fmaxf(r, 0.0f);
                if (RES)  r += (float)rv.h[v];
                o.h[v] = (_Float16)r;
            }
            *(v8h*)((_Float16*)yout + (size_t)pos * OC + ocBase) = o.v;
        }
    }
}

// ---------------------------------------------------------------------------
// e2[k] = sum_c emb[k][c]^2
// ---------------------------------------------------------------------------
__global__ __launch_bounds__(256) void vq_e2_kernel(
    const float* __restrict__ emb, float* __restrict__ e2, int K, int C)
{
    const int k = blockIdx.x * blockDim.x + threadIdx.x;
    if (k < K) {
        float s = 0.0f;
        for (int c = 0; c < C; ++c) { const float v = emb[k * C + c]; s += v * v; }
        e2[k] = s;
    }
}

// ---------------------------------------------------------------------------
// VQ argmin + gather. Writes q in reference NCHW-f32 permuted layout AND a
// NHWC f16 copy for the decoder. cross(z,emb) via WMMA.
// ---------------------------------------------------------------------------
template<int C, int K, int H, int W>
__global__ __launch_bounds__(32) void vq_wmma(
    const float* __restrict__ z, const float* __restrict__ emb,
    const float* __restrict__ e2, float* __restrict__ q,
    _Float16* __restrict__ q16)
{
    constexpr int HW = H * W;
    const int lane = threadIdx.x;
    const int hi   = lane >> 4;
    const int lm   = lane & 15;
    const int p0   = blockIdx.x * 16;

    union V16 { v16h v; _Float16 h[16]; };
    V16 af[C / 32];
    {
        const int pos = p0 + lm;
        const int b   = pos / HW;
        const int off = pos - b * HW;
        const float* zp = z + (size_t)b * C * HW + off;
        #pragma unroll
        for (int kc = 0; kc < C / 32; ++kc) {
            #pragma unroll
            for (int e = 0; e < 16; ++e) {
                const int k = hi * 16 + e;
                af[kc].h[e] = (_Float16)zp[(kc * 32 + k) * HW];
            }
        }
    }

    __shared__ float sDist[16][16];
    __shared__ int   sIdx[16];

    float bestv = 3.4e38f;
    int   besti = 0;
    const int kb2 = hi * 16;

    for (int nt = 0; nt < K / 16; ++nt) {
        const int code = nt * 16 + lm;
        const float* ep = emb + (size_t)code * C + kb2;
        v8f acc = {};
        #pragma unroll
        for (int kc = 0; kc < C / 32; ++kc) {
            V16 bf;
            const float4* fp = (const float4*)(ep + kc * 32);
            #pragma unroll
            for (int j = 0; j < 4; ++j) {
                const float4 f = fp[j];
                bf.h[j * 4 + 0] = (_Float16)f.x;
                bf.h[j * 4 + 1] = (_Float16)f.y;
                bf.h[j * 4 + 2] = (_Float16)f.z;
                bf.h[j * 4 + 3] = (_Float16)f.w;
            }
            acc = __builtin_amdgcn_wmma_f32_16x16x32_f16(
                false, af[kc].v, false, bf.v, (short)0, acc, false, false);
        }
        const float e2n = e2[code];
        #pragma unroll
        for (int v = 0; v < 8; ++v)
            sDist[hi * 8 + v][lm] = e2n - 2.0f * acc[v];
        __syncthreads();
        if (lane < 16) {
            #pragma unroll
            for (int j = 0; j < 16; ++j) {
                const float dv = sDist[lane][j];
                if (dv < bestv) { bestv = dv; besti = nt * 16 + j; }
            }
        }
        __syncthreads();
    }
    if (lane < 16) sIdx[lane] = besti;
    __syncthreads();

    // q[b][c][w][h] = emb[idx[b][h][w]][c]; q16 = NHWC copy of q's (w,h) grid
    for (int t = lane; t < 16 * C; t += 32) {
        const int pi = t / C;
        const int c  = t - pi * C;
        const int p2 = p0 + pi;
        const int b  = p2 / HW;
        const int o2 = p2 - b * HW;
        const int h  = o2 / W;
        const int w  = o2 - h * W;
        const float v = emb[(size_t)sIdx[pi] * C + c];
        q[((size_t)(b * C + c) * W + w) * H + h] = v;
        q16[((size_t)(b * W + w) * H + h) * C + c] = (_Float16)v;
    }
}

// ---------------------------------------------------------------------------
extern "C" void kernel_launch(void* const* d_in, const int* in_sizes, int n_in,
                              void* d_out, int out_size, void* d_ws, size_t ws_size,
                              hipStream_t stream)
{
    (void)in_sizes; (void)n_in; (void)out_size; (void)ws_size;
    const float* x       = (const float*)d_in[0];
    const float* enc_w1  = (const float*)d_in[1];
    const float* enc_b1  = (const float*)d_in[2];
    const float* enc_w2  = (const float*)d_in[3];
    const float* enc_b2  = (const float*)d_in[4];
    const float* enc_w3  = (const float*)d_in[5];
    const float* enc_b3  = (const float*)d_in[6];
    const float* rb1_w1  = (const float*)d_in[7];
    const float* rb1_b1  = (const float*)d_in[8];
    const float* rb1_w2  = (const float*)d_in[9];
    const float* rb1_b2  = (const float*)d_in[10];
    const float* rb2_w1  = (const float*)d_in[11];
    const float* rb2_b1  = (const float*)d_in[12];
    const float* rb2_w2  = (const float*)d_in[13];
    const float* rb2_b2  = (const float*)d_in[14];
    const float* emb     = (const float*)d_in[15];
    const float* dec_w1  = (const float*)d_in[16];
    const float* dec_b1  = (const float*)d_in[17];
    const float* drb1_w1 = (const float*)d_in[18];
    const float* drb1_b1 = (const float*)d_in[19];
    const float* drb1_w2 = (const float*)d_in[20];
    const float* drb1_b2 = (const float*)d_in[21];
    const float* drb2_w1 = (const float*)d_in[22];
    const float* drb2_b1 = (const float*)d_in[23];
    const float* drb2_w2 = (const float*)d_in[24];
    const float* drb2_b2 = (const float*)d_in[25];
    const float* dec_tw1 = (const float*)d_in[26];
    const float* dec_tb1 = (const float*)d_in[27];
    const float* dec_tw2 = (const float*)d_in[28];
    const float* dec_tb2 = (const float*)d_in[29];

    const int B = 32, IC = 3, H0 = 128, C = 128, K = 512;
    const int XHAT_N = B * IC * H0 * H0;
    const int Z_N    = B * C * 32 * 32;

    float* out   = (float*)d_out;
    float* x_hat = out;
    float* z_out = out + XHAT_N;
    float* q_out = out + XHAT_N + Z_N;

    // ---- workspace carve (f16 regions, 32B aligned) ----
    _Float16* hp = (_Float16*)d_ws;
    auto alloc = [&](size_t n) {
        _Float16* p = hp;
        hp += ((n + 15) & ~(size_t)15);
        return p;
    };
    _Float16* h1   = alloc((size_t)32 * 64 * 64 * 128);  // 64x64 NHWC
    _Float16* hbig = alloc((size_t)32 * 64 * 64 * 128);  // 64x64 NHWC
    _Float16* h2   = alloc((size_t)32 * 32 * 32 * 128);
    _Float16* h3   = alloc((size_t)32 * 32 * 32 * 128);
    _Float16* h4   = alloc((size_t)32 * 32 * 32 * 128);
    _Float16* h5   = alloc((size_t)32 * 32 * 32 * 128);
    _Float16* q16  = alloc((size_t)32 * 32 * 32 * 128);
    _Float16* w1h  = alloc(128 * 16 * 3);
    _Float16* w2h  = alloc(128 * 16 * 128);
    _Float16* w3h  = alloc(128 * 9 * 128);
    _Float16* r1ah = alloc(128 * 9 * 128);
    _Float16* r1bh = alloc(128 * 128);
    _Float16* r2ah = alloc(128 * 9 * 128);
    _Float16* r2bh = alloc(128 * 128);
    _Float16* dw1h = alloc(128 * 9 * 128);
    _Float16* d1ah = alloc(128 * 9 * 128);
    _Float16* d1bh = alloc(128 * 128);
    _Float16* d2ah = alloc(128 * 9 * 128);
    _Float16* d2bh = alloc(128 * 128);
    _Float16* tw1h = alloc(128 * 16 * 128);
    _Float16* tw2h = alloc(3 * 16 * 128);
    float* e2 = (float*)(((uintptr_t)hp + 63) & ~(uintptr_t)63);

    const dim3 blk(256);

    // ---- weight prep (f32 -> f16, K-order (kh,kw,ic)) ----
    prep_w<3,128,4,4,false><<<dim3(24),   blk, 0, stream>>>(enc_w1, w1h);
    prep_w<128,128,4,4,false><<<dim3(1024), blk, 0, stream>>>(enc_w2, w2h);
    prep_w<128,128,3,3,false><<<dim3(576),  blk, 0, stream>>>(enc_w3, w3h);
    prep_w<128,128,3,3,false><<<dim3(576),  blk, 0, stream>>>(rb1_w1, r1ah);
    prep_w<128,128,1,1,false><<<dim3(64),   blk, 0, stream>>>(rb1_w2, r1bh);
    prep_w<128,128,3,3,false><<<dim3(576),  blk, 0, stream>>>(rb2_w1, r2ah);
    prep_w<128,128,1,1,false><<<dim3(64),   blk, 0, stream>>>(rb2_w2, r2bh);
    prep_w<128,128,3,3,false><<<dim3(576),  blk, 0, stream>>>(dec_w1, dw1h);
    prep_w<128,128,3,3,false><<<dim3(576),  blk, 0, stream>>>(drb1_w1, d1ah);
    prep_w<128,128,1,1,false><<<dim3(64),   blk, 0, stream>>>(drb1_w2, d1bh);
    prep_w<128,128,3,3,false><<<dim3(576),  blk, 0, stream>>>(drb2_w1, d2ah);
    prep_w<128,128,1,1,false><<<dim3(64),   blk, 0, stream>>>(drb2_w2, d2bh);
    prep_w<128,128,4,4,true><<<dim3(1024), blk, 0, stream>>>(dec_tw1, tw1h);
    prep_w<128,3,4,4,true><<<dim3(24),    blk, 0, stream>>>(dec_tw2, tw2h);

    // ---- encoder ----
    conv_wmma<3,128,128,128,64,64,4,4,2,1, FLAG_RELU_OUT | FLAG_IN_NCHW32>
        <<<dim3(32*64*64/128), blk, 0, stream>>>(x, w1h, enc_b1, nullptr, h1);
    conv_wmma<128,128,64,64,32,32,4,4,2,1, FLAG_RELU_OUT>
        <<<dim3(32*32*32/128), blk, 0, stream>>>(h1, w2h, enc_b2, nullptr, h2);
    conv_wmma<128,128,32,32,32,32,3,3,1,1, 0>
        <<<dim3(32*32*32/128), blk, 0, stream>>>(h2, w3h, enc_b3, nullptr, h3);
    conv_wmma<128,128,32,32,32,32,3,3,1,1, FLAG_RELU_IN>
        <<<dim3(32*32*32/128), blk, 0, stream>>>(h3, r1ah, rb1_b1, nullptr, h4);
    conv_wmma<128,128,32,32,32,32,1,1,1,0, FLAG_RELU_IN | FLAG_RESID>
        <<<dim3(32*32*32/128), blk, 0, stream>>>(h4, r1bh, rb1_b2, h3, h5);
    conv_wmma<128,128,32,32,32,32,3,3,1,1, FLAG_RELU_IN>
        <<<dim3(32*32*32/128), blk, 0, stream>>>(h5, r2ah, rb2_b1, nullptr, h4);
    // z written in reference NCHW f32 layout straight into d_out
    conv_wmma<128,128,32,32,32,32,1,1,1,0,
              FLAG_RELU_IN | FLAG_RESID | FLAG_OUT_NCHW32>
        <<<dim3(32*32*32/128), blk, 0, stream>>>(h4, r2bh, rb2_b2, h5, z_out);

    // ---- vector quantization ----
    vq_e2_kernel<<<dim3((K + 255) / 256), blk, 0, stream>>>(emb, e2, K, C);
    vq_wmma<128, 512, 32, 32>
        <<<dim3(32*32*32/16), dim3(32), 0, stream>>>(z_out, emb, e2, q_out, q16);

    // ---- decoder (input = q16, NHWC f16 copy of the STE forward value) ----
    conv_wmma<128,128,32,32,32,32,3,3,1,1, 0>
        <<<dim3(32*32*32/128), blk, 0, stream>>>(q16, dw1h, dec_b1, nullptr, h3);
    conv_wmma<128,128,32,32,32,32,3,3,1,1, FLAG_RELU_IN>
        <<<dim3(32*32*32/128), blk, 0, stream>>>(h3, d1ah, drb1_b1, nullptr, h4);
    conv_wmma<128,128,32,32,32,32,1,1,1,0, FLAG_RELU_IN | FLAG_RESID>
        <<<dim3(32*32*32/128), blk, 0, stream>>>(h4, d1bh, drb1_b2, h3, h5);
    conv_wmma<128,128,32,32,32,32,3,3,1,1, FLAG_RELU_IN>
        <<<dim3(32*32*32/128), blk, 0, stream>>>(h5, d2ah, drb2_b1, nullptr, h4);
    conv_wmma<128,128,32,32,32,32,1,1,1,0, FLAG_RELU_IN | FLAG_RESID>
        <<<dim3(32*32*32/128), blk, 0, stream>>>(h4, d2bh, drb2_b2, h5, h2);
    conv_wmma<128,128,32,32,64,64,4,4,2,1, FLAG_TRANSPOSE | FLAG_RELU_OUT>
        <<<dim3(32*64*64/128), blk, 0, stream>>>(h2, tw1h, dec_tb1, nullptr, hbig);
    conv_wmma<128,3,64,64,128,128,4,4,2,1, FLAG_TRANSPOSE | FLAG_OUT_NCHW32>
        <<<dim3(32*128*128/128), blk, 0, stream>>>(hbig, tw2h, dec_tb2, nullptr, x_hat);
}
